// GIN0_87909390614643
// MI455X (gfx1250) — matrix operand order repaired and checked by reading, not compiled, not run on hardware
//
#include <hip/hip_runtime.h>
#include <hip/hip_bf16.h>

#define NN 100000
#define EE 1600000
#define GG 128
#define HH 64
#define LL 7

typedef __attribute__((ext_vector_type(16))) _Float16 v16h;
typedef __attribute__((ext_vector_type(8)))  _Float16 v8h;
typedef __attribute__((ext_vector_type(2)))  _Float16 v2h;
typedef __attribute__((ext_vector_type(8)))  float    v8f;

// ---------------- utility kernels ----------------

__global__ void zero_f32_kernel(float* __restrict__ p, int n) {
    int i = blockIdx.x * blockDim.x + threadIdx.x;
    if (i < n) p[i] = 0.0f;
}

// Convert 15 [64x64] weight matrices (k-major) to f16 transposed Wt[m][n][k].
// m=0: cW2 ; m=1..7: W1s[i] ; m=8..14: W2s[i]
__global__ void convert_weights_kernel(const float* __restrict__ cW2,
                                       const float* __restrict__ W1s,
                                       const float* __restrict__ W2s,
                                       _Float16* __restrict__ Wt) {
    int idx = blockIdx.x * blockDim.x + threadIdx.x;
    if (idx >= 15 * 4096) return;
    int m = idx >> 12, e = idx & 4095;
    int n = e >> 6, k = e & 63;
    float v;
    if (m == 0)      v = cW2[k * 64 + n];
    else if (m <= 7) v = W1s[(m - 1) * 4096 + k * 64 + n];
    else             v = W2s[(m - 8) * 4096 + k * 64 + n];
    Wt[m * 4096 + n * 64 + k] = (_Float16)v;
}

// Fold eval-mode BN into scale/shift: s = g*rsqrt(v+eps), t = be - m*s.
// slot0: conv bn1, slot1: conv bn2, slot 2+2i: layer-i bn1, slot 3+2i: layer-i bn2
__global__ void bn_fold_kernel(const float* cg1, const float* cbe1, const float* cm1, const float* cv1,
                               const float* cg2, const float* cbe2, const float* cm2, const float* cv2,
                               const float* g1s, const float* be1s, const float* m1s, const float* v1s,
                               const float* g2s, const float* be2s, const float* m2s, const float* v2s,
                               float* __restrict__ foldS, float* __restrict__ foldT) {
    int idx = blockIdx.x * blockDim.x + threadIdx.x;
    if (idx >= 16 * 64) return;
    int slot = idx >> 6, j = idx & 63;
    float g, be, m, v;
    if (slot == 0)      { g = cg1[j]; be = cbe1[j]; m = cm1[j]; v = cv1[j]; }
    else if (slot == 1) { g = cg2[j]; be = cbe2[j]; m = cm2[j]; v = cv2[j]; }
    else {
        int i = (slot - 2) >> 1;
        if (((slot - 2) & 1) == 0) { g = g1s[i*64+j]; be = be1s[i*64+j]; m = m1s[i*64+j]; v = v1s[i*64+j]; }
        else                       { g = g2s[i*64+j]; be = be2s[i*64+j]; m = m2s[i*64+j]; v = v2s[i*64+j]; }
    }
    float s = g * rsqrtf(v + 1e-5f);
    foldS[idx] = s;
    foldT[idx] = be - m * s;
}

// ---------------- aggregation (scatter-add) ----------------

// conv1: scalar features, 1 thread per edge
__global__ void scatter_scalar_kernel(const float* __restrict__ x,
                                      const int* __restrict__ src,
                                      const int* __restrict__ dst,
                                      float* __restrict__ agg0) {
    int e = blockIdx.x * blockDim.x + threadIdx.x;
    if (e >= EE) return;
    float v = x[src[e]];
    (void)__hip_atomic_fetch_add(agg0 + dst[e], v, __ATOMIC_RELAXED, __HIP_MEMORY_SCOPE_AGENT);
}

// H=64 features: 16 threads per edge, float4 per thread (L2-resident gathers)
__global__ void scatter_edges_kernel(const float* __restrict__ h,
                                     const int* __restrict__ src,
                                     const int* __restrict__ dst,
                                     float* __restrict__ agg) {
    int gid = blockIdx.x * blockDim.x + threadIdx.x;
    int e = gid >> 4;
    if (e >= EE) return;
    int sub = gid & 15;
    int s = src[e], d = dst[e];
    const float4 v = *(const float4*)(h + (size_t)s * 64 + sub * 4);
    float* p = agg + (size_t)d * 64 + sub * 4;
    (void)__hip_atomic_fetch_add(p + 0, v.x, __ATOMIC_RELAXED, __HIP_MEMORY_SCOPE_AGENT);
    (void)__hip_atomic_fetch_add(p + 1, v.y, __ATOMIC_RELAXED, __HIP_MEMORY_SCOPE_AGENT);
    (void)__hip_atomic_fetch_add(p + 2, v.z, __ATOMIC_RELAXED, __HIP_MEMORY_SCOPE_AGENT);
    (void)__hip_atomic_fetch_add(p + 3, v.w, __ATOMIC_RELAXED, __HIP_MEMORY_SCOPE_AGENT);
}

// ---------------- WMMA tile GEMM (16 nodes x 64 -> 64) ----------------
// ldsIn: wave-private f16 tile [16][64]. Wt: f16 [n][k] (transposed weights).
// Epilogue: y = max(acc + bias[n], 0) * scl[n] + shf[n].
// TO_LDS=true: write f16 intermediate back to LDS. TO_LDS=false: write f32 rows to gOut.
template <bool TO_LDS>
__device__ __forceinline__ void wave_tile_gemm(const _Float16* ldsIn,
                                               const _Float16* __restrict__ Wt,
                                               const float* __restrict__ bias,
                                               const float* __restrict__ scl,
                                               const float* __restrict__ shf,
                                               _Float16* ldsOut,
                                               float* gOut,
                                               bool valid) {
    const int lane = threadIdx.x & 31;
    const int row  = lane & 15;
    const int grp  = lane >> 4;

    // A fragments (16-bit A 16x32) in this code's contiguous k order:
    // elements 0..15 -> K = grp*16 + 0..15 (contiguous, matches B)
    v8h x0 = *(const v8h*)(ldsIn + row * 64 +  0 + grp * 16);
    v8h x1 = *(const v8h*)(ldsIn + row * 64 +  8 + grp * 16);
    v8h x2 = *(const v8h*)(ldsIn + row * 64 + 32 + grp * 16);
    v8h x3 = *(const v8h*)(ldsIn + row * 64 + 40 + grp * 16);
    v16h aLo = __builtin_shufflevector(x0, x1, 0,1,2,3,4,5,6,7,8,9,10,11,12,13,14,15);
    v16h aHi = __builtin_shufflevector(x2, x3, 0,1,2,3,4,5,6,7,8,9,10,11,12,13,14,15);

    // 8 back-to-back WMMAs into 4 column-tile accumulators
    v8f acc[4];
#pragma unroll
    for (int ct = 0; ct < 4; ++ct) {
        const int n = ct * 16 + row;
        // B 32x16 layout: lane holds column n, elements = 16 consecutive K (grp selects K half)
        v16h b0 = *(const v16h*)(Wt + n * 64 +  0 + grp * 16);   // K 0..31
        v16h b1 = *(const v16h*)(Wt + n * 64 + 32 + grp * 16);   // K 32..63
        v8f c = {};
        c = __builtin_amdgcn_wmma_f32_16x16x32_f16(false, aLo, false, b0, (short)0, c, false, false);
        c = __builtin_amdgcn_wmma_f32_16x16x32_f16(false, aHi, false, b1, (short)0, c, false, false);
        acc[ct] = c;
    }

    // straight-line epilogue
    if (TO_LDS) {
#pragma unroll
        for (int ct = 0; ct < 4; ++ct) {
            const int n = ct * 16 + row;
            const float bb = bias[n], ss = scl[n], tt = shf[n];
#pragma unroll
            for (int r = 0; r < 8; ++r) {
                float y = acc[ct][r] + bb;
                y = (y > 0.0f) ? y : 0.0f;
                y = y * ss + tt;
                ldsOut[(r + grp * 8) * 64 + n] = (_Float16)y;
            }
        }
    } else {
        if (valid) {
#pragma unroll
            for (int ct = 0; ct < 4; ++ct) {
                const int n = ct * 16 + row;
                const float bb = bias[n], ss = scl[n], tt = shf[n];
#pragma unroll
                for (int r = 0; r < 8; ++r) {
                    float y = acc[ct][r] + bb;
                    y = (y > 0.0f) ? y : 0.0f;
                    y = y * ss + tt;
                    gOut[(size_t)(r + grp * 8) * 64 + n] = y;
                }
            }
        }
    }
}

// ---------------- GIN layer (H -> H), in-place capable ----------------
__global__ __launch_bounds__(256) void gin_layer_kernel(
        const float* __restrict__ h, const float* __restrict__ agg,
        const _Float16* __restrict__ Wt1, const float* __restrict__ b1,
        const float* __restrict__ s1, const float* __restrict__ t1,
        const _Float16* __restrict__ Wt2, const float* __restrict__ b2,
        const float* __restrict__ s2, const float* __restrict__ t2,
        float* __restrict__ hout) {
    __shared__ _Float16 smem[8 * 16 * 64];
    const int wave = threadIdx.x >> 5;
    const int lane = threadIdx.x & 31;
    const int tile = blockIdx.x * 8 + wave;
    const int node0 = tile * 16;
    const bool valid = node0 < NN;
    _Float16* lds = smem + wave * (16 * 64);

    // phase 0: input tile = h + agg, f32 -> f16 (2 cols per iteration, b32 DS stores)
    for (int idx = lane; idx < 512; idx += 32) {
        int r = idx >> 5, cp = (idx & 31) * 2;
        int nd = node0 + r; if (nd >= NN) nd = NN - 1;
        const float2 hv = *(const float2*)(h   + (size_t)nd * 64 + cp);
        const float2 av = *(const float2*)(agg + (size_t)nd * 64 + cp);
        v2h pk;
        pk[0] = (_Float16)(hv.x + av.x);
        pk[1] = (_Float16)(hv.y + av.y);
        *(v2h*)(lds + r * 64 + cp) = pk;
    }
    __syncthreads();
    // phase 1: linear1 + relu + bn1 -> LDS f16
    wave_tile_gemm<true>(lds, Wt1, b1, s1, t1, lds, nullptr, valid);
    __syncthreads();
    // phase 2: linear2 + relu + bn2 -> global f32 (in place)
    wave_tile_gemm<false>(lds, Wt2, b2, s2, t2, nullptr, hout + (size_t)node0 * 64, valid);
}

// ---------------- conv1: Linear(1,64)+relu+bn then WMMA linear2 ----------------
__global__ __launch_bounds__(256) void conv1_kernel(
        const float* __restrict__ x, const float* __restrict__ agg0,
        const float* __restrict__ cW1, const float* __restrict__ cb1,
        const float* __restrict__ s1, const float* __restrict__ t1,
        const _Float16* __restrict__ Wt2, const float* __restrict__ b2,
        const float* __restrict__ s2, const float* __restrict__ t2,
        float* __restrict__ hout) {
    __shared__ _Float16 smem[8 * 16 * 64];
    const int wave = threadIdx.x >> 5;
    const int lane = threadIdx.x & 31;
    const int tile = blockIdx.x * 8 + wave;
    const int node0 = tile * 16;
    const bool valid = node0 < NN;
    _Float16* lds = smem + wave * (16 * 64);

    // phase 0: rank-1 first linear + relu + bn1 (VALU), 2 cols per iteration
    for (int idx = lane; idx < 512; idx += 32) {
        int r = idx >> 5, cp = (idx & 31) * 2;
        int nd = node0 + r; if (nd >= NN) nd = NN - 1;
        float sv = x[nd] + agg0[nd];
        float y0 = sv * cW1[cp + 0] + cb1[cp + 0];
        float y1 = sv * cW1[cp + 1] + cb1[cp + 1];
        y0 = (y0 > 0.0f) ? y0 : 0.0f;
        y1 = (y1 > 0.0f) ? y1 : 0.0f;
        y0 = y0 * s1[cp + 0] + t1[cp + 0];
        y1 = y1 * s1[cp + 1] + t1[cp + 1];
        v2h pk; pk[0] = (_Float16)y0; pk[1] = (_Float16)y1;
        *(v2h*)(lds + r * 64 + cp) = pk;
    }
    __syncthreads();
    wave_tile_gemm<false>(lds, Wt2, b2, s2, t2, nullptr, hout + (size_t)node0 * 64, valid);
}

// ---------------- pooling + head ----------------

__global__ void pool_kernel(const float* __restrict__ h, const int* __restrict__ batch,
                            float* __restrict__ pooled, float* __restrict__ cnt) {
    int idx = blockIdx.x * blockDim.x + threadIdx.x;
    if (idx >= NN * 64) return;
    int node = idx >> 6, col = idx & 63;
    int b = batch[node];
    (void)__hip_atomic_fetch_add(pooled + b * 64 + col, h[idx], __ATOMIC_RELAXED, __HIP_MEMORY_SCOPE_AGENT);
    if (col == 0)
        (void)__hip_atomic_fetch_add(cnt + b, 1.0f, __ATOMIC_RELAXED, __HIP_MEMORY_SCOPE_AGENT);
}

__global__ __launch_bounds__(128) void head_kernel(
        const float* __restrict__ pooled, const float* __restrict__ cnt,
        const float* __restrict__ l1W, const float* __restrict__ l1b,
        const float* __restrict__ l2W, const float* __restrict__ l2b,
        float* __restrict__ out) {
    __shared__ float zs[GG * 64];
    int g = threadIdx.x;
    if (g < GG) {
        float c = cnt[g];
        float inv = 1.0f / (c > 1.0f ? c : 1.0f);
        for (int j = 0; j < 64; ++j) {
            float a = l1b[j];
            for (int k = 0; k < 64; ++k)
                a += pooled[g * 64 + k] * inv * l1W[k * 64 + j];
            zs[g * 64 + j] = (a > 0.0f) ? a : 0.0f;
        }
        float o0 = l2b[0], o1 = l2b[1], o2 = l2b[2];
        for (int k = 0; k < 64; ++k) {
            float z = zs[g * 64 + k];
            o0 += z * l2W[k * 3 + 0];
            o1 += z * l2W[k * 3 + 1];
            o2 += z * l2W[k * 3 + 2];
        }
        float mx = fmaxf(o0, fmaxf(o1, o2));
        float se = __expf(o0 - mx) + __expf(o1 - mx) + __expf(o2 - mx);
        float lse = mx + __logf(se);
        out[g * 3 + 0] = o0 - lse;
        out[g * 3 + 1] = o1 - lse;
        out[g * 3 + 2] = o2 - lse;
    }
}

// ---------------- launch ----------------

extern "C" void kernel_launch(void* const* d_in, const int* in_sizes, int n_in,
                              void* d_out, int out_size, void* d_ws, size_t ws_size,
                              hipStream_t stream) {
    const float* x    = (const float*)d_in[0];
    const int*   ei   = (const int*)d_in[1];
    const int*   src  = ei;
    const int*   dst  = ei + EE;
    const int*   batch = (const int*)d_in[2];
    const float* cW1 = (const float*)d_in[3];
    const float* cb1 = (const float*)d_in[4];
    const float* cg1 = (const float*)d_in[5];
    const float* cbe1 = (const float*)d_in[6];
    const float* cm1 = (const float*)d_in[7];
    const float* cv1 = (const float*)d_in[8];
    const float* cW2 = (const float*)d_in[9];
    const float* cb2 = (const float*)d_in[10];
    const float* cg2 = (const float*)d_in[11];
    const float* cbe2 = (const float*)d_in[12];
    const float* cm2 = (const float*)d_in[13];
    const float* cv2 = (const float*)d_in[14];
    const float* W1s = (const float*)d_in[15];
    const float* b1s = (const float*)d_in[16];
    const float* g1s = (const float*)d_in[17];
    const float* be1s = (const float*)d_in[18];
    const float* m1s = (const float*)d_in[19];
    const float* v1s = (const float*)d_in[20];
    const float* W2s = (const float*)d_in[21];
    const float* b2s = (const float*)d_in[22];
    const float* g2s = (const float*)d_in[23];
    const float* be2s = (const float*)d_in[24];
    const float* m2s = (const float*)d_in[25];
    const float* v2s = (const float*)d_in[26];
    const float* l1W = (const float*)d_in[27];
    const float* l1b = (const float*)d_in[28];
    const float* l2W = (const float*)d_in[29];
    const float* l2b = (const float*)d_in[30];

    // workspace layout
    char* ws = (char*)d_ws;
    float*    h     = (float*)ws;                                  // N*64 f32
    float*    agg   = h + (size_t)NN * 64;                         // N*64 f32 (first N reused as agg0)
    _Float16* Wt    = (_Float16*)(agg + (size_t)NN * 64);          // 15*4096 f16
    float*    foldS = (float*)(Wt + 15 * 4096);                    // 16*64
    float*    foldT = foldS + 16 * 64;                             // 16*64
    float*    pooled = foldT + 16 * 64;                            // G*64
    float*    cnt    = pooled + GG * 64;                           // G

    // 1. weight convert/transpose + BN fold
    convert_weights_kernel<<<(15 * 4096 + 255) / 256, 256, 0, stream>>>(cW2, W1s, W2s, Wt);
    bn_fold_kernel<<<4, 256, 0, stream>>>(cg1, cbe1, cm1, cv1, cg2, cbe2, cm2, cv2,
                                          g1s, be1s, m1s, v1s, g2s, be2s, m2s, v2s,
                                          foldS, foldT);

    const int tiles = (NN + 15) / 16;            // 6250
    const int gemmBlocks = (tiles + 7) / 8;      // 782

    // 2. conv1: scalar aggregation + MLP
    zero_f32_kernel<<<(NN + 255) / 256, 256, 0, stream>>>(agg, NN);
    scatter_scalar_kernel<<<(EE + 255) / 256, 256, 0, stream>>>(x, src, dst, agg);
    conv1_kernel<<<gemmBlocks, 256, 0, stream>>>(x, agg, cW1, cb1,
                                                 foldS + 0 * 64, foldT + 0 * 64,
                                                 Wt + 0 * 4096, cb2,
                                                 foldS + 1 * 64, foldT + 1 * 64, h);

    // 3. stacked GIN layers (in place on h)
    for (int i = 0; i < LL; ++i) {
        zero_f32_kernel<<<(NN * 64 + 255) / 256, 256, 0, stream>>>(agg, NN * 64);
        scatter_edges_kernel<<<(EE * 16 + 255) / 256, 256, 0, stream>>>(h, src, dst, agg);
        gin_layer_kernel<<<gemmBlocks, 256, 0, stream>>>(
            h, agg,
            Wt + (1 + i) * 4096, b1s + i * 64, foldS + (2 + 2 * i) * 64, foldT + (2 + 2 * i) * 64,
            Wt + (8 + i) * 4096, b2s + i * 64, foldS + (3 + 2 * i) * 64, foldT + (3 + 2 * i) * 64,
            h);
    }

    // 4. mean pool + head
    zero_f32_kernel<<<(GG * 64 + GG + 255) / 256, 256, 0, stream>>>(pooled, GG * 64 + GG);
    pool_kernel<<<(NN * 64 + 255) / 256, 256, 0, stream>>>(h, batch, pooled, cnt);
    head_kernel<<<1, 128, 0, stream>>>(pooled, cnt, l1W, l1b, l2W, l2b, (float*)d_out);
}
